// MyRNN_16527034155805
// MI455X (gfx1250) — matrix operand order repaired and checked by reading, not compiled, not run on hardware
//
#include <hip/hip_runtime.h>
#include <stdint.h>

// Problem constants (from reference)
#define B_   1024
#define T_   80
#define E_   100
#define U_   512
#define G_   1536   // 3U
#define EP_  128    // E padded to multiple of 32 (WMMA K)

// Tiling
#define BM   16            // batch rows per workgroup
#define NWG  (B_/BM)       // 64 workgroups
#define NTHR 512           // 16 waves (wave32); each wave owns 2 U-tiles
#define XSTR 136           // x LDS row stride (bf16 elems): 16B aligned, +4 banks/row
#define HSTR 520           // h LDS row stride (bf16 elems): 16B aligned, +4 banks/row
#define BMH  (BM * HSTR)   // one h buffer, in elems

typedef __attribute__((ext_vector_type(16))) __bf16 v16bf;
typedef __attribute__((ext_vector_type(8)))  float  v8f;

union V16 { uint4 q[2]; v16bf v; };

__device__ __forceinline__ unsigned short f2bf(float f) {
    unsigned int u = __float_as_uint(f);
    u = (u + 0x7FFFu + ((u >> 16) & 1u)) >> 16;   // round-to-nearest-even
    return (unsigned short)u;
}
__device__ __forceinline__ float bf2f(unsigned short h) {
    return __uint_as_float(((unsigned int)h) << 16);
}
__device__ __forceinline__ float sigmoidf_(float x) {
    return 1.0f / (1.0f + __expf(-x));
}
__device__ __forceinline__ float tanhf_(float x) {
    x = fminf(15.0f, fmaxf(-15.0f, x));
    float e = __expf(2.0f * x);
    return (e - 1.0f) / (e + 1.0f);
}
__device__ __forceinline__ v8f vzero() {
    v8f z = {0.f,0.f,0.f,0.f,0.f,0.f,0.f,0.f};
    return z;
}

// Opaque zero in an SGPR: added to a weight pointer inside the time loop it makes
// the loads loop-variant (blocks LICM -> no hoist -> no scratch spill) while
// keeping pointer provenance intact so global_load (not flat_load) is selected.
__device__ __forceinline__ int opaque_zero() {
    int v = 0;
    asm volatile("" : "+s"(v));
    return v;
}

// One GEMM phase: accumulate A(16xK) x {Wz,Wr,Wh}(Kx16) into 3 accumulators.
// abase: LDS activation pointer (direct GEP on a __shared__ array at the call
//        site so ds_load_b128 is emitted), pre-offset by m*stride + c0.
//        A VGPR layout: lanes 0-15 hold K {0..15}; lanes 16-31
//        hold {16..31}.
// wbase: swizzled bf16 weight pointer for the z-gate column tile of this U-tile,
//        pre-offset by lane*16; r/h gate tiles sit at constant element offsets
//        GOF/2*GOF which fold into global-load immediates.
__device__ __forceinline__ void gru_mm(
    const unsigned short* abase,
    const unsigned short* __restrict__ wbase,
    int nK, v8f& az, v8f& ar, v8f& ah)
{
    const size_t GOF = (size_t)32 * (size_t)nK * 512;  // elems between gate blocks
#pragma unroll 4
    for (int kt = 0; kt < nK; ++kt) {
        V16 A, Bz, Br, Bh;
        const uint4* ap = (const uint4*)(abase + kt * 32);
        A.q[0] = ap[0];            // K chunk c0..c0+7
        A.q[1] = ap[1];            // K chunk c0+8..c0+15
        const uint4* pz = (const uint4*)(wbase + (size_t)kt * 512);
        Bz.q[0] = pz[0]; Bz.q[1] = pz[1];
        const uint4* pr = (const uint4*)(wbase + GOF + (size_t)kt * 512);
        Br.q[0] = pr[0]; Br.q[1] = pr[1];
        const uint4* ph = (const uint4*)(wbase + 2 * GOF + (size_t)kt * 512);
        Bh.q[0] = ph[0]; Bh.q[1] = ph[1];
        az = __builtin_amdgcn_wmma_f32_16x16x32_bf16(false, A.v, false, Bz.v, (short)0, az, false, false);
        ar = __builtin_amdgcn_wmma_f32_16x16x32_bf16(false, A.v, false, Br.v, (short)0, ar, false, false);
        ah = __builtin_amdgcn_wmma_f32_16x16x32_bf16(false, A.v, false, Bh.v, (short)0, ah, false, false);
    }
}

// Convert fp32 weights -> bf16, swizzled so B-matrix loads are 2x global_load_b128
// per lane per 32x16 tile. Tile (nt,kt) stored at (nt*nK + kt)*512; within a tile,
// lane L owns 16 contiguous elems:
//   L<16 :  W[kt*32 + 0..15,  nt*16 + L]
//   L>=16:  W[kt*32 + 16..31, nt*16 + (L-16)]
// k0 is K-padded from 100 to 128 with zero rows.
__global__ void swizzle_weights(const float* __restrict__ k0,
                                const float* __restrict__ rk0,
                                const float* __restrict__ k1,
                                const float* __restrict__ rk1,
                                unsigned short* __restrict__ ws)
{
    const size_t S0 = (size_t)EP_ * G_;
    const size_t S  = (size_t)U_  * G_;
    const size_t total = S0 + 3 * S;
    size_t idx = (size_t)blockIdx.x * blockDim.x + threadIdx.x;
    if (idx >= total) return;

    const float* src; int Kp, Kr; size_t e;
    if (idx < S0)            { src = k0;  Kp = EP_; Kr = E_; e = idx; }
    else if (idx < S0 + S)   { src = rk0; Kp = U_;  Kr = U_; e = idx - S0; }
    else if (idx < S0 + 2*S) { src = k1;  Kp = U_;  Kr = U_; e = idx - S0 - S; }
    else                     { src = rk1; Kp = U_;  Kr = U_; e = idx - S0 - 2*S; }

    int    nK     = Kp >> 5;
    size_t tile   = e >> 9;
    int    within = (int)(e & 511);
    int    lane   = within >> 4;
    int    j      = within & 15;
    int    nt     = (int)(tile / (size_t)nK);
    int    kt     = (int)(tile % (size_t)nK);
    int    k      = kt * 32 + ((lane >> 4) << 4) + j;
    int    n      = nt * 16 + (lane & 15);
    float  v      = (k < Kr) ? src[(size_t)k * G_ + n] : 0.0f;
    ws[idx] = f2bf(v);
}

__global__ __launch_bounds__(NTHR)
void gru_persistent(const int*   __restrict__ tokens,
                    const float* __restrict__ emb,
                    const float* __restrict__ b0,
                    const float* __restrict__ b1,
                    const float* __restrict__ w_out,
                    const float* __restrict__ b_out,
                    const unsigned short* __restrict__ ws,
                    float* __restrict__ out)
{
    // double-buffered h state (bf16, flat 1-D so buffer select stays a GEP) + x stage
    __shared__ __align__(16) unsigned short h0s[2 * BMH];
    __shared__ __align__(16) unsigned short h1s[2 * BMH];
    __shared__ __align__(16) unsigned short xs [BM * XSTR];
    __shared__ int   toks[BM];
    __shared__ float red[NTHR];

    const int tid   = threadIdx.x;
    const int lane  = tid & 31;
    const int w     = tid >> 5;              // wave id (0..15); owns U-tiles w, w+16
    const int brow0 = blockIdx.x * BM;

    // zero-init LDS (h state = 0; x padding cols [100,128) stay 0 forever)
    for (int i = tid; i < 2 * BMH; i += NTHR) { h0s[i] = 0; h1s[i] = 0; }
    for (int i = tid; i < BM * XSTR; i += NTHR) xs[i] = 0;

    // Per-lane bias scalars for both owned U-tiles (loaded once).
    float bz0[2], br0[2], bhx0[2], bhh0[2], bz1[2], br1[2], bhx1[2], bhh1[2];
    int   colw[2];
#pragma unroll
    for (int s = 0; s < 2; ++s) {
        const int cz = (w + 16 * s) * 16 + (lane & 15);
        colw[s] = cz;
        bz0[s]  = b0[cz]          + b0[G_ + cz];
        br0[s]  = b0[U_ + cz]     + b0[G_ + U_ + cz];
        bhx0[s] = b0[2*U_ + cz];
        bhh0[s] = b0[G_ + 2*U_ + cz];
        bz1[s]  = b1[cz]          + b1[G_ + cz];
        br1[s]  = b1[U_ + cz]     + b1[G_ + U_ + cz];
        bhx1[s] = b1[2*U_ + cz];
        bhh1[s] = b1[G_ + 2*U_ + cz];
    }

    // Swizzled weight base pointers: z-gate tile of each (matrix, U-tile), per lane.
    const size_t S0 = (size_t)EP_ * G_;
    const size_t S  = (size_t)U_  * G_;
    const int loff = lane * 16;
    const unsigned short* wk0[2];
    const unsigned short* wr0[2];
    const unsigned short* wk1[2];
    const unsigned short* wr1[2];
#pragma unroll
    for (int s = 0; s < 2; ++s) {
        const size_t ut = (size_t)(w + 16 * s);
        wk0[s] = ws                 + (ut * 4 ) * 512 + loff;
        wr0[s] = ws + S0            + (ut * 16) * 512 + loff;
        wk1[s] = ws + S0 + S        + (ut * 16) * 512 + loff;
        wr1[s] = ws + S0 + 2 * S    + (ut * 16) * 512 + loff;
    }

    // A-matrix LDS element offsets (contiguous-k A layout used here: m = lane&15, c0 = 16 for hi half)
    const int m    = lane & 15;
    const int c0   = (lane >> 4) << 4;
    const int aoff = m * HSTR + c0;          // within one h buffer
    const int hi8  = (lane >> 4) << 3;       // C/D layout: lanes>=16 hold rows M+8

    v8f h0r[2] = { vzero(), vzero() };       // owned h tiles, fp32, C-layout
    v8f h1r[2] = { vzero(), vzero() };

    __syncthreads();

    for (int t = 0; t < T_; ++t) {
        const int cur = (t & 1) * BMH, nxt = cur ^ BMH;

        // ---- gather x(t) = emb[tokens[:, t]] into LDS (bf16) ----
        if (tid < BM) toks[tid] = tokens[(brow0 + tid) * T_ + t];
        __syncthreads();
        for (int i = tid; i < BM * E_; i += NTHR) {
            int row = i / E_;
            int col = i - row * E_;
            xs[row * XSTR + col] = f2bf(emb[(size_t)toks[row] * E_ + col]);
        }
        __syncthreads();

        // Per-step opaque zero: keeps the small (nK=4) k0-phase weight loads
        // inside the loop (re-read from L2-hot ws) instead of hoist+spill,
        // without losing global address-space inference.
        const int oz = opaque_zero();

        // ---- layer 0: gates = x@k0 + h0@rk0 ; write h0(t) into other buffer ----
#pragma unroll
        for (int s = 0; s < 2; ++s) {
            v8f az = vzero(), ar = vzero(), ahx = vzero(), ahh = vzero();
            gru_mm(&xs[m * XSTR + c0], wk0[s] + oz, EP_ / 32, az, ar, ahx);
            gru_mm(&h0s[cur + aoff],   wr0[s],      U_  / 32, az, ar, ahh);
#pragma unroll
            for (int j = 0; j < 8; ++j) {
                float z  = sigmoidf_(az[j] + bz0[s]);
                float r  = sigmoidf_(ar[j] + br0[s]);
                float hc = tanhf_((ahx[j] + bhx0[s]) + r * (ahh[j] + bhh0[s]));
                float hn = z * h0r[s][j] + (1.0f - z) * hc;
                h0r[s][j] = hn;
                h0s[nxt + (hi8 + j) * HSTR + colw[s]] = f2bf(hn);
            }
        }
        __syncthreads();   // h0s[nxt] complete; reads of h0s[cur]/xs finished

        // ---- layer 1: gates = h0(t)@k1 + h1@rk1 ; write h1(t) into other buffer ----
#pragma unroll
        for (int s = 0; s < 2; ++s) {
            v8f az = vzero(), ar = vzero(), ahx = vzero(), ahh = vzero();
            gru_mm(&h0s[nxt + aoff], wk1[s], U_ / 32, az, ar, ahx);
            gru_mm(&h1s[cur + aoff], wr1[s], U_ / 32, az, ar, ahh);
#pragma unroll
            for (int j = 0; j < 8; ++j) {
                float z  = sigmoidf_(az[j] + bz1[s]);
                float r  = sigmoidf_(ar[j] + br1[s]);
                float hc = tanhf_((ahx[j] + bhx1[s]) + r * (ahh[j] + bhh1[s]));
                float hn = z * h1r[s][j] + (1.0f - z) * hc;
                h1r[s][j] = hn;
                h1s[nxt + (hi8 + j) * HSTR + colw[s]] = f2bf(hn);
            }
        }
        __syncthreads();   // h1s[nxt] complete; reads of h1s[cur] finished
    }

    // ---- output: sigmoid(h1 @ w_out + b_out), one scalar per batch row ----
    {
        const int finb = (T_ & 1) * BMH;   // buffer holding h1(T-1)
        int row = tid >> 5;                // 0..15
        int c   = lane;                    // 0..31, 16 elems each
        float p = 0.0f;
#pragma unroll
        for (int j = 0; j < 16; ++j) {
            p += bf2f(h1s[finb + row * HSTR + c * 16 + j]) * w_out[c * 16 + j];
        }
        red[tid] = p;
        __syncthreads();
        if (tid < BM) {
            float sAcc = b_out[0];
            for (int c2 = 0; c2 < 32; ++c2) sAcc += red[tid * 32 + c2];
            out[brow0 + tid] = sigmoidf_(sAcc);
        }
    }
}

extern "C" void kernel_launch(void* const* d_in, const int* in_sizes, int n_in,
                              void* d_out, int out_size, void* d_ws, size_t ws_size,
                              hipStream_t stream)
{
    const int*   tokens = (const int*)  d_in[0];
    const float* emb    = (const float*)d_in[1];
    const float* k0     = (const float*)d_in[2];
    const float* rk0    = (const float*)d_in[3];
    const float* b0     = (const float*)d_in[4];
    const float* k1     = (const float*)d_in[5];
    const float* rk1    = (const float*)d_in[6];
    const float* b1     = (const float*)d_in[7];
    const float* w_out  = (const float*)d_in[8];
    const float* b_out  = (const float*)d_in[9];
    float* out = (float*)d_out;
    unsigned short* ws = (unsigned short*)d_ws;

    // 1) one-shot fp32 -> swizzled bf16 weight conversion into workspace (~5.1 MB)
    const size_t total = (size_t)EP_ * G_ + 3 * (size_t)U_ * G_;
    const int thr = 256;
    const int blk = (int)((total + thr - 1) / thr);
    hipLaunchKernelGGL(swizzle_weights, dim3(blk), dim3(thr), 0, stream,
                       k0, rk0, k1, rk1, ws);

    // 2) persistent GRU scan: 64 WGs x 512 threads (16 wave32 waves, 2 U-tiles each)
    hipLaunchKernelGGL(gru_persistent, dim3(NWG), dim3(NTHR), 0, stream,
                       tokens, emb, b0, b1, w_out, b_out, ws, out);
}
